// MixtureBlock_52063593562963
// MI455X (gfx1250) — matrix operand order repaired and checked by reading, not compiled, not run on hardware
//
#include <hip/hip_runtime.h>
#include <hip/hip_bf16.h>
#include <math.h>

typedef unsigned short u16;
typedef unsigned int   u32;

typedef __bf16 v16bf __attribute__((ext_vector_type(16)));
typedef float  v8f   __attribute__((ext_vector_type(8)));
typedef int    v4i   __attribute__((ext_vector_type(4)));

union FragBF { v16bf v; uint4 u[2]; };

__device__ inline u16 f2bf(float f) {
  union { __bf16 b; u16 u; } x; x.b = (__bf16)f; return x.u;
}

// ---------- CDNA5 async global->LDS copy (16B per lane), guarded ----------
__device__ inline void cp16(const void* g, void* l) {
#if defined(__gfx1250__) && __has_builtin(__builtin_amdgcn_global_load_async_to_lds_b128)
  __builtin_amdgcn_global_load_async_to_lds_b128(
      (__attribute__((address_space(1))) v4i*)g,
      (__attribute__((address_space(3))) v4i*)l, 0, 0);
#else
  *(uint4*)l = *(const uint4*)g;   // synchronous fallback
#endif
}

template <int N> __device__ inline void wait_async() {
#if defined(__gfx1250__) && __has_builtin(__builtin_amdgcn_s_wait_asynccnt)
  __builtin_amdgcn_s_wait_asynccnt((unsigned short)N);
#elif defined(__gfx1250__)
  asm volatile("s_wait_asynccnt %0" ::"n"(N) : "memory");
#endif
}

// =====================================================================
// Generic batched GEMM:  C[z] = act( A[z] (MxK, bf16, row-major)
//                                  x Bt[z]^T (Bt is NxK bf16 row-major)
//                                  + bias )
// Block tile 128x128, 8 waves (4x2), wave tile 32x64 (2x4 WMMA frags),
// K-step 32, double-buffered async LDS staging with XOR 16B swizzle.
// =====================================================================
__global__ __launch_bounds__(256) void gemm_bf16_kernel(
    const u16* __restrict__ A, const u16* __restrict__ Bt,
    const float* __restrict__ bias, void* __restrict__ Cv,
    int M, int N, int K,
    long long sA, long long sB, long long sC,
    int act, int outbf)
{
  __shared__ __align__(16) char smem[2 * 16384];   // 2 buffers x (A 8KB + B 8KB)
  const int tid = threadIdx.x;
  const u16* Ab = A + (size_t)blockIdx.z * (size_t)sA;
  const u16* Bb = Bt + (size_t)blockIdx.z * (size_t)sB;
  const int m0 = blockIdx.y * 128;
  const int n0 = blockIdx.x * 128;
  const int nk = K >> 5;

  auto issue = [&](int kt, int buf) {
    char* la = smem + buf * 16384;
    char* lb = la + 8192;
#pragma unroll
    for (int i = 0; i < 2; ++i) {
      int idx = tid + i * 256;        // 512 chunks of 16B per tile
      int row = idx >> 2;
      int ch  = idx & 3;
      u32 off = (u32)row * 64u + (u32)((ch ^ (row & 3)) << 4);
      cp16(Ab + (size_t)(m0 + row) * K + kt * 32 + ch * 8, la + off);
      cp16(Bb + (size_t)(n0 + row) * K + kt * 32 + ch * 8, lb + off);
    }
  };

  v8f acc[2][4];
#pragma unroll
  for (int i = 0; i < 2; ++i)
#pragma unroll
    for (int j = 0; j < 4; ++j) acc[i][j] = 0.f;

  const int wave = tid >> 5, lane = tid & 31;
  const int wm = (wave >> 1) * 32, wn = (wave & 1) * 64;
  const u32 lhalf = (u32)(lane >> 4), lrow = (u32)(lane & 15);

  issue(0, 0);
  for (int kt = 0; kt < nk; ++kt) {
    if (kt + 1 < nk) { issue(kt + 1, (kt + 1) & 1); wait_async<4>(); }
    else             { wait_async<0>(); }
    __syncthreads();
    const char* la = smem + (kt & 1) * 16384;
    const char* lb = la + 8192;

    FragBF af[2], bfr[4];
#pragma unroll
    for (int i = 0; i < 2; ++i) {
      u32 row = (u32)(wm + i * 16) + lrow;
      // A 16x32 layout: lanes0-15 hold K0..7 & 16..23; lanes16-31 K8..15 & 24..31
      af[i].u[0] = *(const uint4*)(la + row * 64u + (((lhalf)     ^ (row & 3)) << 4));
      af[i].u[1] = *(const uint4*)(la + row * 64u + (((lhalf + 2) ^ (row & 3)) << 4));
    }
#pragma unroll
    for (int j = 0; j < 4; ++j) {
      u32 row = (u32)(wn + j * 16) + lrow;
      // B (N-major): lanes0-15 col n, K0..7 & 16..23; lanes16-31 col n, K8..15 & 24..31
      bfr[j].u[0] = *(const uint4*)(lb + row * 64u + (((lhalf)     ^ (row & 3)) << 4));
      bfr[j].u[1] = *(const uint4*)(lb + row * 64u + (((lhalf + 2) ^ (row & 3)) << 4));
    }
#pragma unroll
    for (int i = 0; i < 2; ++i)
#pragma unroll
      for (int j = 0; j < 4; ++j)
        acc[i][j] = __builtin_amdgcn_wmma_f32_16x16x32_bf16(
            false, af[i].v, false, bfr[j].v, (short)0, acc[i][j], false, false);
    __syncthreads();
  }

  float* Cf = (float*)Cv + (size_t)blockIdx.z * (size_t)sC;
  u16*   Cb = (u16*)Cv  + (size_t)blockIdx.z * (size_t)sC;
#pragma unroll
  for (int i = 0; i < 2; ++i)
#pragma unroll
    for (int j = 0; j < 4; ++j) {
      int col = n0 + wn + j * 16 + (int)lrow;
      float bsv = bias ? bias[col] : 0.f;
#pragma unroll
      for (int r = 0; r < 8; ++r) {
        int rowm = m0 + wm + i * 16 + (int)lhalf * 8 + r;   // C layout: VGPR r -> M=r / r+8
        float c = acc[i][j][r] + bsv;
        if (act) c = 0.5f * c * (1.f + erff(c * 0.70710678118654752f));  // exact GELU
        size_t o = (size_t)rowm * N + col;
        if (outbf) Cb[o] = f2bf(c); else Cf[o] = c;
      }
    }
}

// ---------------- f32 -> bf16 bulk convert (4 elems/thread) ----------------
__global__ __launch_bounds__(256) void cvt_bf16_kernel(const float* __restrict__ in,
                                                       u16* __restrict__ out) {
  size_t i = ((((size_t)blockIdx.x << 8) + threadIdx.x) << 2);
  float4 v = *(const float4*)(in + i);
  ushort4 o; o.x = f2bf(v.x); o.y = f2bf(v.y); o.z = f2bf(v.z); o.w = f2bf(v.w);
  *(ushort4*)(out + i) = o;
}

// ------------- W (KxN f32) -> Wt (NxK bf16) tiled transpose -------------
__global__ void transpose_cvt_kernel(const float* __restrict__ W, u16* __restrict__ Wt,
                                     int K, int N) {
  __shared__ float tile[32][33];
  const int k0 = blockIdx.y << 5, n0 = blockIdx.x << 5;
  const int tx = threadIdx.x, ty = threadIdx.y;   // (32,8)
#pragma unroll
  for (int i = 0; i < 4; ++i)
    tile[ty + 8 * i][tx] = W[(size_t)(k0 + ty + 8 * i) * N + n0 + tx];
  __syncthreads();
#pragma unroll
  for (int i = 0; i < 4; ++i)
    Wt[(size_t)(n0 + ty + 8 * i) * K + k0 + tx] = f2bf(tile[tx][ty + 8 * i]);
}

// ------- per-(b,l,h) row: optional L2-normalize, relayout head-major, bf16 -------
// One wave32 per 64-elem row; 2 floats/lane, shuffle reduction.
__global__ __launch_bounds__(256) void pack_kernel(const float* __restrict__ attn,
                                                   u16* __restrict__ Kb,
                                                   u16* __restrict__ Qb,
                                                   int normalize) {
  const int wave = threadIdx.x >> 5, lane = threadIdx.x & 31;
  const int gid = blockIdx.x * 8 + wave;            // (b, l, h) row id
  const int b = gid >> 15, rem = gid & 32767, l = rem >> 4, h = rem & 15;
  const float2 v = *(const float2*)(attn + (((size_t)(b * 2048 + l)) << 10) + h * 64 + lane * 2);
  float ss = v.x * v.x + v.y * v.y;
#pragma unroll
  for (int m = 16; m >= 1; m >>= 1) ss += __shfl_xor(ss, m, 32);
  float sc = normalize ? (1.f / fmaxf(sqrtf(ss), 1e-12f)) : 1.f;
  u16* dst = (l < 1024)
                 ? (Kb + ((((size_t)(b * 16 + h)) << 10) + l) * 64)
                 : (Qb + ((((size_t)(b * 16 + h)) << 10) + (l - 1024)) * 64);
  ushort2 o; o.x = f2bf(v.x * sc); o.y = f2bf(v.y * sc);
  *(ushort2*)(dst + lane * 2) = o;
}

// ---- per-row 0.3 quantile of 1024 scores: radix-select ranks 306 & 307 ----
__global__ __launch_bounds__(256) void quantile_kernel(const float* __restrict__ sc,
                                                       float* __restrict__ thr) {
  __shared__ u32 skey[1024];
  __shared__ u32 hist[256];
  __shared__ u32 s_prefix;
  __shared__ int s_rank;
  __shared__ u32 s_res[2];
  const int tid = threadIdx.x;
  const size_t base = (size_t)blockIdx.x << 10;
  for (int i = tid; i < 1024; i += 256) {
    u32 u = __float_as_uint(sc[base + i]);
    skey[i] = (u & 0x80000000u) ? ~u : (u | 0x80000000u);   // order-preserving key
  }
  __syncthreads();
  for (int r = 0; r < 2; ++r) {
    if (tid == 0) { s_prefix = 0; s_rank = 306 + r; }
    __syncthreads();
    for (int shift = 24; shift >= 0; shift -= 8) {
      hist[tid & 255] = 0;
      __syncthreads();
      u32 mask = (shift == 24) ? 0u : (0xFFFFFFFFu << (shift + 8));
      u32 pfx = s_prefix;
      for (int i = tid; i < 1024; i += 256) {
        u32 k = skey[i];
        if ((k & mask) == pfx) atomicAdd(&hist[(k >> shift) & 255u], 1u);
      }
      __syncthreads();
      if (tid == 0) {
        u32 cum = 0;
        for (int d = 0; d < 256; ++d) {
          u32 c = hist[d];
          if (cum + c > (u32)s_rank) { s_prefix |= (u32)d << shift; s_rank -= (int)cum; break; }
          cum += c;
        }
      }
      __syncthreads();
    }
    if (tid == 0) s_res[r] = s_prefix;
    __syncthreads();
  }
  if (tid == 0) {
    u32 a = s_res[0], c = s_res[1];
    u32 ba = (a & 0x80000000u) ? (a ^ 0x80000000u) : ~a;
    u32 bc = (c & 0x80000000u) ? (c ^ 0x80000000u) : ~c;
    // jnp.quantile linear: pos = 0.3*1023 = 306.9
    thr[blockIdx.x] = 0.1f * __uint_as_float(ba) + 0.9f * __uint_as_float(bc);
  }
}

// ---- fused: softmax(logits/8) * gate, renorm (Z cancels), mean over heads ----
__global__ __launch_bounds__(256) void fused_kernel(const float* __restrict__ scL,
                                                    const float* __restrict__ scG,
                                                    const float* __restrict__ thr,
                                                    float* __restrict__ out) {
  __shared__ float red[256];
  const int tid = threadIdx.x;
  const int b = blockIdx.x >> 10, q = blockIdx.x & 1023;
  const float scale = 0.125f;   // 64^-0.5
  float a0 = 0.f, a1 = 0.f, a2 = 0.f, a3 = 0.f;
  for (int h = 0; h < 16; ++h) {
    const size_t base = ((((size_t)(b * 16 + h)) << 10) + q) << 10;
    float4 lg = *(const float4*)(scL + base + tid * 4);
    float4 gs = *(const float4*)(scG + base + tid * 4);
    float tv = thr[((b * 16 + h) << 10) + q];
    red[tid] = fmaxf(fmaxf(lg.x, lg.y), fmaxf(lg.z, lg.w));
    __syncthreads();
    for (int s = 128; s > 0; s >>= 1) { if (tid < s) red[tid] = fmaxf(red[tid], red[tid + s]); __syncthreads(); }
    float Mx = red[0] * scale;
    __syncthreads();
    float e0 = (gs.x >= tv) ? __expf(lg.x * scale - Mx) : 0.f;
    float e1 = (gs.y >= tv) ? __expf(lg.y * scale - Mx) : 0.f;
    float e2 = (gs.z >= tv) ? __expf(lg.z * scale - Mx) : 0.f;
    float e3 = (gs.w >= tv) ? __expf(lg.w * scale - Mx) : 0.f;
    red[tid] = e0 + e1 + e2 + e3;
    __syncthreads();
    for (int s = 128; s > 0; s >>= 1) { if (tid < s) red[tid] += red[tid + s]; __syncthreads(); }
    float rinv = 1.f / red[0];
    __syncthreads();
    a0 += e0 * rinv; a1 += e1 * rinv; a2 += e2 * rinv; a3 += e3 * rinv;
  }
  const float inv = 1.f / 16.f;
  float4 o; o.x = a0 * inv; o.y = a1 * inv; o.z = a2 * inv; o.w = a3 * inv;
  *(float4*)(out + (((size_t)blockIdx.x) << 10) + tid * 4) = o;
}

extern "C" void kernel_launch(void* const* d_in, const int* in_sizes, int n_in,
                              void* d_out, int out_size, void* d_ws, size_t ws_size,
                              hipStream_t stream) {
  (void)in_sizes; (void)n_in; (void)out_size; (void)ws_size;
  const float* hidden = (const float*)d_in[0];
  const float* Wg1 = (const float*)d_in[1];
  const float* bg1 = (const float*)d_in[2];
  const float* Wg2 = (const float*)d_in[3];
  const float* bg2 = (const float*)d_in[4];
  const float* Wl1 = (const float*)d_in[5];
  const float* bl1 = (const float*)d_in[6];
  const float* Wl2 = (const float*)d_in[7];
  const float* bl2 = (const float*)d_in[8];
  float* out = (float*)d_out;

  const int B = 4, L = 2048, D = 1024, F = 4096, H = 16, S = 1024;
  const int M = B * L;  // 8192

  char* ws = (char*)d_ws;
  size_t off = 0;
  auto carve = [&](size_t bytes) { char* p = ws + off; off = (off + bytes + 255) & ~(size_t)255; return p; };

  u16*  Xbf  = (u16*)carve((size_t)M * D * 2);
  u16*  W1gt = (u16*)carve((size_t)F * D * 2);
  u16*  W2gt = (u16*)carve((size_t)D * F * 2);
  u16*  W1lt = (u16*)carve((size_t)F * D * 2);
  u16*  W2lt = (u16*)carve((size_t)D * F * 2);
  u16*  Hb   = (u16*)carve((size_t)M * F * 2);
  float* attn = (float*)carve((size_t)M * D * 4);
  u16*  Kg = (u16*)carve((size_t)B * H * S * 64 * 2);
  u16*  Qg = (u16*)carve((size_t)B * H * S * 64 * 2);
  u16*  Kl = (u16*)carve((size_t)B * H * S * 64 * 2);
  u16*  Ql = (u16*)carve((size_t)B * H * S * 64 * 2);
  float* scG = (float*)carve((size_t)B * H * S * S * 4);
  float* scL = (float*)carve((size_t)B * H * S * S * 4);
  float* thr = (float*)carve((size_t)B * H * S * 4);

  // precision staging
  cvt_bf16_kernel<<<(M * D) / 1024, 256, 0, stream>>>(hidden, Xbf);
  dim3 tb(32, 8);
  transpose_cvt_kernel<<<dim3(F / 32, D / 32), tb, 0, stream>>>(Wg1, W1gt, D, F);
  transpose_cvt_kernel<<<dim3(D / 32, F / 32), tb, 0, stream>>>(Wg2, W2gt, F, D);
  transpose_cvt_kernel<<<dim3(F / 32, D / 32), tb, 0, stream>>>(Wl1, W1lt, D, F);
  transpose_cvt_kernel<<<dim3(D / 32, F / 32), tb, 0, stream>>>(Wl2, W2lt, F, D);

  // gates FFN
  gemm_bf16_kernel<<<dim3(F / 128, M / 128, 1), 256, 0, stream>>>(
      Xbf, W1gt, bg1, Hb, M, F, D, 0, 0, 0, /*gelu*/1, /*bf16*/1);
  gemm_bf16_kernel<<<dim3(D / 128, M / 128, 1), 256, 0, stream>>>(
      Hb, W2gt, bg2, attn, M, D, F, 0, 0, 0, 0, 0);
  pack_kernel<<<(B * L * H) / 8, 256, 0, stream>>>(attn, Kg, Qg, /*normalize*/1);

  // logits FFN (reuses H and attn buffers)
  gemm_bf16_kernel<<<dim3(F / 128, M / 128, 1), 256, 0, stream>>>(
      Xbf, W1lt, bl1, Hb, M, F, D, 1, 1, 1, 1, 1);
  gemm_bf16_kernel<<<dim3(D / 128, M / 128, 1), 256, 0, stream>>>(
      Hb, W2lt, bl2, attn, M, D, F, 0, 0, 0, 0, 0);
  pack_kernel<<<(B * L * H) / 8, 256, 0, stream>>>(attn, Kl, Ql, 0);

  // 64-batch score GEMMs: (1024x64) x (64x1024)
  long long sQK = (long long)S * 64, sSc = (long long)S * S;
  gemm_bf16_kernel<<<dim3(S / 128, S / 128, B * H), 256, 0, stream>>>(
      Qg, Kg, nullptr, scG, S, S, 64, sQK, sQK, sSc, 0, 0);
  gemm_bf16_kernel<<<dim3(S / 128, S / 128, B * H), 256, 0, stream>>>(
      Ql, Kl, nullptr, scL, S, S, 64, sQK, sQK, sSc, 0, 0);

  // per-row quantile thresholds, then fused softmax/gate/renorm/head-mean
  quantile_kernel<<<B * H * S, 256, 0, stream>>>(scG, thr);
  fused_kernel<<<B * S, 256, 0, stream>>>(scL, scG, thr, out);
}
